// Attention_84602265796942
// MI455X (gfx1250) — compile-verified, matrix operand order checked by reading, not run on hardware
//
#include <hip/hip_runtime.h>
#include <hip/hip_bf16.h>

// ---------------------------------------------------------------------------
// Types
// ---------------------------------------------------------------------------
typedef __attribute__((ext_vector_type(16))) __bf16 v16bf;
typedef __attribute__((ext_vector_type(8)))  float  v8f;

struct alignas(16) U16B { unsigned int w[4]; };   // POD 16-byte chunk

union Frag {
  v16bf v;
  U16B  q[2];
};

__device__ __forceinline__ v8f wmma_bf16(v16bf a, v16bf b, v8f c) {
  // D = A(16x32 bf16) x B(32x16 bf16) + C(16x16 f32)
  return __builtin_amdgcn_wmma_f32_16x16x32_bf16(
      /*neg_a=*/false, a, /*neg_b=*/false, b,
      /*c_mod=*/(short)0, c, /*reuse_a=*/false, /*reuse_b=*/false);
}

// ---------------------------------------------------------------------------
// DPP butterfly reduction within each 16-lane row (no LDS round trips)
// ---------------------------------------------------------------------------
template<int CTRL>
__device__ __forceinline__ float dpp_xchg(float x) {
  int xi = __builtin_bit_cast(int, x);
  int r  = __builtin_amdgcn_update_dpp(0, xi, CTRL, 0xF, 0xF, true);
  return __builtin_bit_cast(float, r);
}

__device__ __forceinline__ float row16_max(float x) {
  x = fmaxf(x, dpp_xchg<0xB1>(x));    // quad_perm [1,0,3,2]  (xor 1)
  x = fmaxf(x, dpp_xchg<0x4E>(x));    // quad_perm [2,3,0,1]  (xor 2)
  x = fmaxf(x, dpp_xchg<0x141>(x));   // row_half_mirror      (xor 7)
  x = fmaxf(x, dpp_xchg<0x140>(x));   // row_mirror           (xor 15)
  return x;
}

__device__ __forceinline__ float row16_sum(float x) {
  x += dpp_xchg<0xB1>(x);
  x += dpp_xchg<0x4E>(x);
  x += dpp_xchg<0x141>(x);
  x += dpp_xchg<0x140>(x);
  return x;
}

// ---------------------------------------------------------------------------
// CDNA5 async global->LDS copy (guarded; falls back to plain load/store)
// Builtin pointer params are int-__vector(4)* with AS1 / AS3 qualifiers.
// ---------------------------------------------------------------------------
#ifndef USE_ASYNC_COPY
#if __has_builtin(__builtin_amdgcn_global_load_async_to_lds_b128) && \
    __has_builtin(__builtin_amdgcn_s_wait_asynccnt)
#define USE_ASYNC_COPY 1
#else
#define USE_ASYNC_COPY 0
#endif
#endif

#if USE_ASYNC_COPY
typedef __attribute__((ext_vector_type(4))) int v4i_t;
typedef __attribute__((address_space(1))) v4i_t gv4i_t;
typedef __attribute__((address_space(3))) v4i_t lv4i_t;
#endif

__device__ __forceinline__ void cp16_g2l(void* lds, const void* g) {
#if USE_ASYNC_COPY
  __builtin_amdgcn_global_load_async_to_lds_b128(
      (gv4i_t*)(void*)g, (lv4i_t*)lds, 0, 0);
#else
  *(U16B*)lds = *(const U16B*)g;
#endif
}

__device__ __forceinline__ void lds_fence_barrier() {
#if USE_ASYNC_COPY
  __builtin_amdgcn_s_wait_asynccnt(0);
#endif
  __syncthreads();
}

// ---------------------------------------------------------------------------
// fp32 -> bf16 conversion
// ---------------------------------------------------------------------------
__global__ void cvt_f32_bf16(const float* __restrict__ s, __bf16* __restrict__ d, int n) {
  int stride = gridDim.x * blockDim.x;
  for (int i = blockIdx.x * blockDim.x + threadIdx.x; i < n; i += stride)
    d[i] = (__bf16)s[i];
}

// ---------------------------------------------------------------------------
// C[M,N] = A[M,K] @ B[N,K]^T   (A,B bf16 row-major, K-contiguous)
// block tile 128x256 (8 waves, 64x64 per wave), kc=64,
// double-buffered async global->LDS pipeline, WMMA bf16 f32-acc.
// ---------------------------------------------------------------------------
#define GA_ELEMS (128 * 72)   // one A buffer (elements, +8 row pad)
#define GB_ELEMS (256 * 72)   // one B buffer

__device__ __forceinline__ void issue_tile(const __bf16* __restrict__ A,
                                           const __bf16* __restrict__ Bm,
                                           __bf16* As, __bf16* Bs,
                                           int tid, int mBase, int nBase,
                                           int K, int k0)
{
  // A tile: 128 rows x 64 cols = 1024 16B chunks (4 per thread)
#pragma unroll
  for (int i = 0; i < 4; i++) {
    int ch = tid + i * 256;
    int r = ch >> 3, c = (ch & 7) * 8;
    cp16_g2l(&As[r * 72 + c], A + (size_t)(mBase + r) * K + k0 + c);
  }
  // B tile: 256 rows x 64 cols = 2048 chunks (8 per thread)
#pragma unroll
  for (int i = 0; i < 8; i++) {
    int ch = tid + i * 256;
    int r = ch >> 3, c = (ch & 7) * 8;
    cp16_g2l(&Bs[r * 72 + c], Bm + (size_t)(nBase + r) * K + k0 + c);
  }
}

template<bool OUT_F32>
__global__ __launch_bounds__(256, 1)
void gemm_xwt(const __bf16* __restrict__ A, const __bf16* __restrict__ Bm,
              __bf16* __restrict__ Cb, float* __restrict__ Cf,
              int M, int N, int K)
{
  extern __shared__ __align__(16) char gsm[];
  __bf16* AsBase = (__bf16*)gsm;              // [2][128][72]
  __bf16* BsBase = AsBase + 2 * GA_ELEMS;     // [2][256][72]

  const int tid  = threadIdx.x;
  const int wave = tid >> 5;
  const int lane = tid & 31;
  const int hf   = lane >> 4;      // K-half select (A/B fragment layout)
  const int l    = lane & 15;
  const int wm   = wave >> 2;      // 0..1  (M direction)
  const int wn   = wave & 3;       // 0..3  (N direction)

  const int mBase = blockIdx.y * 128;
  const int nBase = blockIdx.x * 256;

  const v8f vz = {0.f,0.f,0.f,0.f,0.f,0.f,0.f,0.f};
  v8f acc[4][4];
#pragma unroll
  for (int i = 0; i < 4; i++)
#pragma unroll
    for (int j = 0; j < 4; j++) acc[i][j] = vz;

  // prime the pipeline
  issue_tile(A, Bm, AsBase, BsBase, tid, mBase, nBase, K, 0);

  const int nk = K >> 6;
  for (int kt = 0; kt < nk; kt++) {
    const __bf16* Asc = AsBase + (kt & 1) * GA_ELEMS;
    const __bf16* Bsc = BsBase + (kt & 1) * GB_ELEMS;

    // copies for tile kt were issued one iteration ago -> latency hidden
    lds_fence_barrier();

    if (kt + 1 < nk)
      issue_tile(A, Bm,
                 AsBase + ((kt + 1) & 1) * GA_ELEMS,
                 BsBase + ((kt + 1) & 1) * GB_ELEMS,
                 tid, mBase, nBase, K, (kt + 1) << 6);
    if (kt + 2 < nk) {   // L2 prefetch two tiles ahead (global_prefetch_b8)
      __builtin_prefetch(A  + (size_t)(mBase + (tid >> 3)) * K + ((kt + 2) << 6), 0, 1);
      __builtin_prefetch(Bm + (size_t)(nBase + (tid >> 3)) * K + ((kt + 2) << 6), 0, 1);
    }

#pragma unroll
    for (int kk = 0; kk < 64; kk += 32) {
      Frag a[4], b[4];
#pragma unroll
      for (int mi = 0; mi < 4; mi++) {
        const __bf16* p = &Asc[(wm * 64 + mi * 16 + l) * 72 + kk + hf * 8];
        a[mi].q[0] = *(const U16B*)p;
        a[mi].q[1] = *(const U16B*)(p + 16);
      }
#pragma unroll
      for (int ni = 0; ni < 4; ni++) {
        const __bf16* p = &Bsc[(wn * 64 + ni * 16 + l) * 72 + kk + hf * 8];
        b[ni].q[0] = *(const U16B*)p;
        b[ni].q[1] = *(const U16B*)(p + 16);
      }
#pragma unroll
      for (int mi = 0; mi < 4; mi++)
#pragma unroll
        for (int ni = 0; ni < 4; ni++)
          acc[mi][ni] = wmma_bf16(a[mi].v, b[ni].v, acc[mi][ni]);
    }
    // no trailing barrier needed: a buffer is rewritten only after the next
    // lds_fence_barrier(), which every wave reaches only after finishing
    // its reads of that buffer.
  }

  // Epilogue: C layout -> VGPR i: M = i (lanes 0-15) / 8+i (lanes 16-31), N = l
#pragma unroll
  for (int mi = 0; mi < 4; mi++)
#pragma unroll
    for (int ni = 0; ni < 4; ni++)
#pragma unroll
      for (int i = 0; i < 8; i++) {
        int r = mBase + wm * 64 + mi * 16 + (hf ? 8 + i : i);
        int c = nBase + wn * 64 + ni * 16 + l;
        float v = acc[mi][ni][i];
        if (OUT_F32) Cf[(size_t)r * N + c] = v;
        else         Cb[(size_t)r * N + c] = (__bf16)v;
      }
}

// ---------------------------------------------------------------------------
// RoPE, in-place on bf16 (S, NH*64); freqs are (S, 32) f32
// ---------------------------------------------------------------------------
__global__ void rope_bf16(__bf16* __restrict__ t, const float* __restrict__ cptr,
                          const float* __restrict__ sptr, int S, int NH)
{
  int total = S * NH * 32;
  int stride = gridDim.x * blockDim.x;
  for (int idx = blockIdx.x * blockDim.x + threadIdx.x; idx < total; idx += stride) {
    int j = idx & 31;
    int h = (idx >> 5) % NH;
    int s = idx / (NH * 32);
    float c  = cptr[s * 32 + j];
    float sn = sptr[s * 32 + j];
    __bf16* p = t + (size_t)s * NH * 64 + h * 64 + 2 * j;
    float a = (float)p[0], b = (float)p[1];
    p[0] = (__bf16)(a * c - b * sn);
    p[1] = (__bf16)(a * sn + b * c);
  }
}

// ---------------------------------------------------------------------------
// Fused flash attention: one block per (batch, q-head, 128-row q tile)
// K (row-major) + V (transposed) for one kv-head staged fully in LDS.
// ---------------------------------------------------------------------------
#define KSTRIDE 72     // 64 + 8 pad (bf16 elements)
#define VSTRIDE 520    // 512 + 8 pad

__global__ __launch_bounds__(256, 1)
void attn_fused(const __bf16* __restrict__ qb, const __bf16* __restrict__ kb,
                const __bf16* __restrict__ vb, __bf16* __restrict__ ob)
{
  extern __shared__ __align__(16) char smem[];
  __bf16* Ksh = (__bf16*)smem;                 // [512][KSTRIDE]  K row-major
  __bf16* Vst = Ksh + 512 * KSTRIDE;           // [64][VSTRIDE]   V transposed
  __bf16* Ps  = Vst + 64 * VSTRIDE;            // [8][16][KSTRIDE] P staging

  const int tid  = threadIdx.x;
  const int wave = tid >> 5;
  const int lane = tid & 31;
  const int hf   = lane >> 4;
  const int l    = lane & 15;

  const int qt = blockIdx.x & 3;               // q tile (128 rows)
  const int hq = (blockIdx.x >> 2) & 31;       // query head
  const int bb = blockIdx.x >> 7;              // batch
  const int g  = hq >> 2;                      // kv head (REP=4)

  // ---- stage K (async b128) and V (transposed, scalar) ----
#pragma unroll 4
  for (int i = 0; i < 16; i++) {
    int ch = tid + i * 256;
    int r = ch >> 3, c = (ch & 7) * 8;
    cp16_g2l(&Ksh[r * KSTRIDE + c], kb + (size_t)(bb * 512 + r) * 512 + g * 64 + c);
  }
#pragma unroll 4
  for (int i = 0; i < 16; i++) {
    int ch = tid + i * 256;
    int r = ch >> 3, c = (ch & 7) * 8;         // key r, dims c..c+7
    U16B d = *(const U16B*)(vb + (size_t)(bb * 512 + r) * 512 + g * 64 + c);
    const __bf16* e = (const __bf16*)&d;
#pragma unroll
    for (int j = 0; j < 8; j++) Vst[(c + j) * VSTRIDE + r] = e[j];
  }
  lds_fence_barrier();

  // ---- per-wave Q fragment: 16 rows x 64 dims ----
  const int qrow0 = qt * 128 + wave * 16;
  const __bf16* qptr = qb + (size_t)(bb * 512 + qrow0 + l) * 2048 + hq * 64;
  Frag qf[2];
#pragma unroll
  for (int ks = 0; ks < 2; ks++) {
    qf[ks].q[0] = *(const U16B*)(qptr + ks * 32 + hf * 8);
    qf[ks].q[1] = *(const U16B*)(qptr + ks * 32 + 16 + hf * 8);
  }

  const v8f vz = {0.f,0.f,0.f,0.f,0.f,0.f,0.f,0.f};
  float mrow[8], lrow[8];
#pragma unroll
  for (int i = 0; i < 8; i++) { mrow[i] = -3.0e38f; lrow[i] = 0.f; }
  v8f oacc[4];
#pragma unroll
  for (int ns = 0; ns < 4; ns++) oacc[ns] = vz;

  __bf16* pw = Ps + wave * 16 * KSTRIDE;
  const float scale = 0.125f;                  // DH^-0.5

  for (int t = 0; t < 8; t++) {                // 8 key tiles of 64
    v8f sacc[4];
#pragma unroll
    for (int ns = 0; ns < 4; ns++) sacc[ns] = vz;

    // S = Q @ K^T  (B-frag lane = key, K-dim = head dim -> contiguous K rows)
#pragma unroll
    for (int ks = 0; ks < 2; ks++)
#pragma unroll
      for (int ns = 0; ns < 4; ns++) {
        Frag kf;
        const __bf16* p = &Ksh[(t * 64 + ns * 16 + l) * KSTRIDE + ks * 32 + hf * 8];
        kf.q[0] = *(const U16B*)p;
        kf.q[1] = *(const U16B*)(p + 16);
        sacc[ns] = wmma_bf16(qf[ks].v, kf.v, sacc[ns]);
      }

    // online softmax (DPP butterfly reductions inside 16-lane halves)
    float alpha[8];
#pragma unroll
    for (int i = 0; i < 8; i++) {
      float mx = fmaxf(fmaxf(sacc[0][i], sacc[1][i]), fmaxf(sacc[2][i], sacc[3][i]));
      mx = row16_max(mx) * scale;
      float mnew = fmaxf(mrow[i], mx);
      alpha[i] = __expf(mrow[i] - mnew);
      mrow[i] = mnew;
      float rs = 0.f;
#pragma unroll
      for (int ns = 0; ns < 4; ns++) {
        float pv = __expf(sacc[ns][i] * scale - mnew);
        sacc[ns][i] = pv;
        rs += pv;
      }
      rs = row16_sum(rs);
      lrow[i] = lrow[i] * alpha[i] + rs;
    }

    // rescale O; stage P (C layout -> A layout via per-wave LDS scratch)
#pragma unroll
    for (int ns = 0; ns < 4; ns++)
#pragma unroll
      for (int i = 0; i < 8; i++) {
        oacc[ns][i] *= alpha[i];
        int pr = hf ? (8 + i) : i;
        pw[pr * KSTRIDE + ns * 16 + l] = (__bf16)sacc[ns][i];
      }

    // O += P @ V  (B-frag from transposed V: lane = dim, K-dim = key)
#pragma unroll
    for (int ks = 0; ks < 2; ks++) {
      Frag pf;
      const __bf16* pp = &pw[l * KSTRIDE + ks * 32 + hf * 8];
      pf.q[0] = *(const U16B*)pp;
      pf.q[1] = *(const U16B*)(pp + 16);
#pragma unroll
      for (int ns = 0; ns < 4; ns++) {
        Frag vf;
        const __bf16* vp = &Vst[(ns * 16 + l) * VSTRIDE + t * 64 + ks * 32 + hf * 8];
        vf.q[0] = *(const U16B*)vp;
        vf.q[1] = *(const U16B*)(vp + 16);
        oacc[ns] = wmma_bf16(pf.v, vf.v, oacc[ns]);
      }
    }
  }

  // normalize + write O (bf16, (S, H*DH))
#pragma unroll
  for (int ns = 0; ns < 4; ns++)
#pragma unroll
    for (int i = 0; i < 8; i++) {
      float inv = 1.f / lrow[i];
      int r = qrow0 + (hf ? 8 + i : i);
      int c = hq * 64 + ns * 16 + l;
      ob[(size_t)(bb * 512 + r) * 2048 + c] = (__bf16)(oacc[ns][i] * inv);
    }
}

// ---------------------------------------------------------------------------
// Host launcher
// ---------------------------------------------------------------------------
extern "C" void kernel_launch(void* const* d_in, const int* in_sizes, int n_in,
                              void* d_out, int out_size, void* d_ws, size_t ws_size,
                              hipStream_t stream)
{
  (void)in_sizes; (void)n_in; (void)out_size; (void)ws_size;

  const float* x  = (const float*)d_in[0];
  const float* fc = (const float*)d_in[1];
  const float* fs = (const float*)d_in[2];
  const float* wq = (const float*)d_in[3];
  const float* wk = (const float*)d_in[4];
  const float* wv = (const float*)d_in[5];
  const float* wo = (const float*)d_in[6];
  float* out = (float*)d_out;

  const int S = 8192, DIM = 2048, H = 32, HKV = 8, DH = 64;

  char* ws = (char*)d_ws;
  size_t off = 0;
  auto alloc = [&](size_t elems) -> __bf16* {
    char* p = ws + off;
    off = (off + elems * 2 + 255) & ~(size_t)255;
    return (__bf16*)p;
  };

  __bf16* xb  = alloc((size_t)S * DIM);          // 33.6 MB
  __bf16* wqb = alloc((size_t)H * DH * DIM);     //  8.4 MB
  __bf16* wkb = alloc((size_t)HKV * DH * DIM);   //  2.1 MB
  __bf16* wvb = alloc((size_t)HKV * DH * DIM);   //  2.1 MB
  __bf16* wob = alloc((size_t)DIM * H * DH);     //  8.4 MB
  __bf16* qbf = alloc((size_t)S * H * DH);       // 33.6 MB
  __bf16* kbf = alloc((size_t)S * HKV * DH);     //  8.4 MB
  __bf16* vbf = alloc((size_t)S * HKV * DH);     //  8.4 MB
  __bf16* obf = alloc((size_t)S * H * DH);       // 33.6 MB

  auto cvt = [&](const float* s, __bf16* d, size_t n) {
    int blocks = (int)((n + 1023) / 1024);
    if (blocks > 2048) blocks = 2048;
    cvt_f32_bf16<<<blocks, 256, 0, stream>>>(s, d, (int)n);
  };
  cvt(x,  xb,  (size_t)S * DIM);
  cvt(wq, wqb, (size_t)H * DH * DIM);
  cvt(wk, wkb, (size_t)HKV * DH * DIM);
  cvt(wv, wvb, (size_t)HKV * DH * DIM);
  cvt(wo, wob, (size_t)DIM * H * DH);

  // GEMM dynamic LDS: 2x(128x72) + 2x(256x72) bf16 = 110592 B
  size_t gemmSmem = (size_t)(2 * GA_ELEMS + 2 * GB_ELEMS) * 2;
  (void)hipFuncSetAttribute((const void*)gemm_xwt<false>,
                            hipFuncAttributeMaxDynamicSharedMemorySize,
                            (int)gemmSmem);
  (void)hipFuncSetAttribute((const void*)gemm_xwt<true>,
                            hipFuncAttributeMaxDynamicSharedMemorySize,
                            (int)gemmSmem);

  dim3 blk(256);
  // Q/K/V projections (bf16 out)
  gemm_xwt<false><<<dim3(2048 / 256, 8192 / 128), blk, gemmSmem, stream>>>(
      xb, wqb, qbf, nullptr, 8192, 2048, 2048);
  gemm_xwt<false><<<dim3(512 / 256, 8192 / 128), blk, gemmSmem, stream>>>(
      xb, wkb, kbf, nullptr, 8192, 512, 2048);
  gemm_xwt<false><<<dim3(512 / 256, 8192 / 128), blk, gemmSmem, stream>>>(
      xb, wvb, vbf, nullptr, 8192, 512, 2048);

  // RoPE on Q and K
  rope_bf16<<<2048, 256, 0, stream>>>(qbf, fc, fs, S, H);
  rope_bf16<<<2048, 256, 0, stream>>>(kbf, fc, fs, S, HKV);

  // Fused attention: grid = B * H * (L/128) = 2048 blocks
  size_t attnSmem = (size_t)(512 * KSTRIDE + 64 * VSTRIDE + 8 * 16 * KSTRIDE) * 2;
  (void)hipFuncSetAttribute((const void*)attn_fused,
                            hipFuncAttributeMaxDynamicSharedMemorySize,
                            (int)attnSmem);
  attn_fused<<<2048, 256, attnSmem, stream>>>(qbf, kbf, vbf, obf);

  // Output projection (f32 out -> d_out)
  gemm_xwt<true><<<dim3(2048 / 256, 8192 / 128), blk, gemmSmem, stream>>>(
      obf, wob, nullptr, out, 8192, 2048, 2048);
}
